// Model_82935818486260
// MI455X (gfx1250) — compile-verified, matrix operand order checked by reading, not run on hardware
//
#include <hip/hip_runtime.h>

typedef __attribute__((ext_vector_type(16))) __bf16 v16bf;
typedef __attribute__((ext_vector_type(8)))  float  v8f;
typedef __attribute__((ext_vector_type(4)))  float  v4f;

union FragAB { v16bf v; v4f q[2]; };
union FragC  { v8f   v; v4f q[2]; };

#define HID      512
#define G4H      2048
#define NTIL     128      // 2048 / 16 N-tiles
#define NKC      16       // 512 / 32 K-chunks
#define TLEN     201
#define NTHREADS 1024     // 32 wave32s: 8 waves/SIMD for L2-latency hiding
#define VTILES   7        // ceil(100/16)

// ---------------------------------------------------------------------------
// Pack f32 row-major weight [nrows x 512] into bf16 WMMA B-fragment layout:
// half index = (((tile*16 + kc)*32 + lane)*16 + group*8 + j)
//   tile : 16 output columns (rows of W, since z = h @ W^T)
//   lane : holds column n = tile*16 + (lane&15); K-base = (lane>=16 ? 8 : 0)
//   group 0: K = kc*32 + kbase + j ; group 1: K = kc*32 + 16 + kbase + j
// Rows >= nrows are zero-padded (used for w_proj, 100 -> 112).
// ---------------------------------------------------------------------------
__global__ void pack_w(const float* __restrict__ W, unsigned short* __restrict__ out,
                       int nrows, int total) {
    int gid = blockIdx.x * blockDim.x + threadIdx.x;
    if (gid >= total) return;
    const int j  = gid & 7;
    const int gr = (gid >> 3) & 1;
    const int l  = (gid >> 4) & 31;
    const int kc = (gid >> 9) & 15;
    const int t  = gid >> 13;
    const int n  = t * 16 + (l & 15);
    const int k  = kc * 32 + gr * 16 + ((l >> 4) << 3) + j;
    float val = (n < nrows) ? W[(size_t)n * HID + k] : 0.0f;
    __bf16 b = (__bf16)val;
    out[gid] = __builtin_bit_cast(unsigned short, b);
}

// ---------------------------------------------------------------------------
// Accumulate C[0..3] += A(16x512 bf16, row-major in LDS) x W-slice.
// Wave `wave` owns one tile per gate: tile(g) = g*32 + wave, g = 0..3,
// i.e. hidden units wave*16 .. wave*16+15 for all four gates.
// ---------------------------------------------------------------------------
__device__ __forceinline__ void mm_accum(const __bf16* srcA, const __bf16* Wpk,
                                         int wave, int lane, v8f C[4]) {
    const int lmod  = lane & 15;
    const int khalf = (lane >> 4) << 3;
#pragma unroll 1
    for (int kc = 0; kc < NKC; ++kc) {
        FragAB A;
        const __bf16* pa = srcA + lmod * HID + kc * 32 + khalf;
        A.q[0] = *(const v4f*)(pa);
        A.q[1] = *(const v4f*)(pa + 16);
        FragAB B[4];
#pragma unroll
        for (int g = 0; g < 4; ++g) {
            const int tile = g * 32 + wave;
            const __bf16* pb = Wpk + (size_t)((tile * NKC + kc) * 32 + lane) * 16;
            B[g].q[0] = *(const v4f*)(pb);
            B[g].q[1] = *(const v4f*)(pb + 8);
        }
#pragma unroll
        for (int g = 0; g < 4; ++g) {
            C[g] = __builtin_amdgcn_wmma_f32_16x16x32_bf16(
                false, A.v, false, B[g].v, (short)0, C[g], false, false);
        }
    }
}

// Gate pre-activations for a frozen input x_t: Xg[t] = x_t @ W_ih^T + b_ih + b_hh,
// stored in C-fragment layout; also archives x_t (bf16 row-major) for projection.
__device__ __forceinline__ void compute_xg(int t, const __bf16* srcA,
                                           const __bf16* wih_pk,
                                           const float* b_ih, const float* b_hh,
                                           __bf16* xbuf, float* xgbuf,
                                           int tid, int wave, int lane) {
    __bf16* xb = xbuf + (size_t)t * (16 * HID);
    for (int idx = tid; idx < 16 * HID; idx += NTHREADS) xb[idx] = srcA[idx];

    const int lmod = lane & 15;
    v8f C[4];
#pragma unroll
    for (int g = 0; g < 4; ++g) {
        const int n = (g * 32 + wave) * 16 + lmod;
        const float bb = b_ih[n] + b_hh[n];
        for (int r = 0; r < 8; ++r) C[g][r] = bb;
    }
    mm_accum(srcA, wih_pk, wave, lane, C);

    float* xg = xgbuf + (size_t)t * (NTIL * 32 * 8);
#pragma unroll
    for (int g = 0; g < 4; ++g) {
        const int tile = g * 32 + wave;
        FragC fc; fc.v = C[g];
        v4f* pg = (v4f*)(xg + (size_t)(tile * 32 + lane) * 8);
        pg[0] = fc.q[0];
        pg[1] = fc.q[1];
    }
}

// One LSTM cell: z = Xg[j] + h@W_hh^T ; gates ; update cstate ; write new h.
__device__ __forceinline__ void lstm_cell(int j, const __bf16* srcA, __bf16* dstA,
                                          const __bf16* whh_pk, const float* xgbuf,
                                          int wave, int lane, v8f& cstate) {
    const int lmod = lane & 15;
    const float* xg = xgbuf + (size_t)j * (NTIL * 32 * 8);
    v8f C[4];
#pragma unroll
    for (int g = 0; g < 4; ++g) {
        const int tile = g * 32 + wave;
        const v4f* pg = (const v4f*)(xg + (size_t)(tile * 32 + lane) * 8);
        FragC fc; fc.q[0] = pg[0]; fc.q[1] = pg[1];
        C[g] = fc.v;
    }
    mm_accum(srcA, whh_pk, wave, lane, C);

    const v8f zi = C[0], zf = C[1], zg = C[2], zo = C[3];
    const int col = wave * 16 + lmod;
#pragma unroll
    for (int r = 0; r < 8; ++r) {
        const float ig = 1.0f / (1.0f + __expf(-zi[r]));
        const float fg = 1.0f / (1.0f + __expf(-zf[r]));
        const float gg = tanhf(zg[r]);
        const float og = 1.0f / (1.0f + __expf(-zo[r]));
        const float cc = fg * cstate[r] + ig * gg;
        cstate[r] = cc;
        const float hh = og * tanhf(cc);
        const int row = (lane < 16) ? r : (r + 8);
        dstA[row * HID + col] = (__bf16)hh;
    }
}

// ---------------------------------------------------------------------------
// Persistent sequential decoder: 1 workgroup, 32 waves, h double-buffered in LDS.
// ---------------------------------------------------------------------------
__global__ __launch_bounds__(NTHREADS, 1) void lstm_main(
    const float* __restrict__ feat, const float* __restrict__ embed,
    const int* __restrict__ sosp,
    const float* __restrict__ b_ih, const float* __restrict__ b_hh,
    const unsigned short* __restrict__ wih_pk_u,
    const unsigned short* __restrict__ whh_pk_u,
    unsigned short* __restrict__ xbuf_u,
    float* xgbuf) {
    __shared__ __bf16 Abuf[2][16 * HID];
    const __bf16* wih_pk = reinterpret_cast<const __bf16*>(wih_pk_u);
    const __bf16* whh_pk = reinterpret_cast<const __bf16*>(whh_pk_u);
    __bf16* xbuf = reinterpret_cast<__bf16*>(xbuf_u);

    const int tid  = threadIdx.x;
    const int lane = tid & 31;
    const int wave = tid >> 5;

    // Abuf[0] = x0 = embed[sos] broadcast over batch rows 0..7 (rows 8..15 zero)
    // Abuf[1] = h0 = feat
    const int sos = *sosp;
    for (int idx = tid; idx < 16 * HID; idx += NTHREADS) {
        const int row = idx >> 9, col = idx & (HID - 1);
        Abuf[0][idx] = (row < 8) ? (__bf16)embed[(size_t)sos * HID + col] : (__bf16)0.0f;
        Abuf[1][idx] = (row < 8) ? (__bf16)feat[(size_t)row * HID + col] : (__bf16)0.0f;
    }
    __syncthreads();

    v8f cstate;
    for (int r = 0; r < 8; ++r) cstate[r] = 0.0f;

    compute_xg(0, &Abuf[0][0], wih_pk, b_ih, b_hh, xbuf, xgbuf, tid, wave, lane);
    __syncthreads();

    int p = 1;  // current h lives in Abuf[p]
#pragma unroll 1
    for (int k = 0; k < TLEN - 1; ++k) {
#pragma unroll 1
        for (int j = 0; j <= k; ++j) {
            lstm_cell(j, &Abuf[p][0], &Abuf[p ^ 1][0], whh_pk, xgbuf, wave, lane, cstate);
            __syncthreads();
            p ^= 1;
        }
        // h after this iteration is x_{k+1}
        compute_xg(k + 1, &Abuf[p][0], wih_pk, b_ih, b_hh, xbuf, xgbuf, tid, wave, lane);
        __syncthreads();
    }
}

// ---------------------------------------------------------------------------
// Projection: res[b, v, t] = x_t[b,:] . w_proj[v,:] + b_proj[v]
// One wave per (t, vtile); 16 WMMAs over K=512.
// ---------------------------------------------------------------------------
__global__ __launch_bounds__(32, 1) void proj_kern(
    const unsigned short* __restrict__ xbuf_u,
    const unsigned short* __restrict__ wpj_u,
    const float* __restrict__ b_proj,
    float* __restrict__ out) {
    const __bf16* xbuf = reinterpret_cast<const __bf16*>(xbuf_u);
    const __bf16* wpj  = reinterpret_cast<const __bf16*>(wpj_u);
    const int t    = blockIdx.x;
    const int vt   = blockIdx.y;
    const int lane = threadIdx.x;
    const int lmod = lane & 15;
    const int khalf = (lane >> 4) << 3;
    const __bf16* xb = xbuf + (size_t)t * (16 * HID);

    v8f C;
    for (int r = 0; r < 8; ++r) C[r] = 0.0f;
#pragma unroll 1
    for (int kc = 0; kc < NKC; ++kc) {
        FragAB A;
        const __bf16* pa = xb + lmod * HID + kc * 32 + khalf;
        A.q[0] = *(const v4f*)(pa);
        A.q[1] = *(const v4f*)(pa + 16);
        FragAB B;
        const __bf16* pb = wpj + (size_t)((vt * NKC + kc) * 32 + lane) * 16;
        B.q[0] = *(const v4f*)(pb);
        B.q[1] = *(const v4f*)(pb + 8);
        C = __builtin_amdgcn_wmma_f32_16x16x32_bf16(
            false, A.v, false, B.v, (short)0, C, false, false);
    }
    if (lane < 16) {
        const int v = vt * 16 + lmod;
        if (v < 100) {
            const float bb = b_proj[v];
#pragma unroll
            for (int r = 0; r < 8; ++r)
                out[((size_t)r * 100 + v) * TLEN + t] = C[r] + bb;
        }
    }
}

extern "C" void kernel_launch(void* const* d_in, const int* in_sizes, int n_in,
                              void* d_out, int out_size, void* d_ws, size_t ws_size,
                              hipStream_t stream) {
    (void)in_sizes; (void)n_in; (void)out_size; (void)ws_size;
    const float* feat   = (const float*)d_in[0];
    const float* w_ih   = (const float*)d_in[1];
    const float* w_hh   = (const float*)d_in[2];
    const float* b_ih   = (const float*)d_in[3];
    const float* b_hh   = (const float*)d_in[4];
    const float* embed  = (const float*)d_in[5];
    const float* w_proj = (const float*)d_in[6];
    const float* b_proj = (const float*)d_in[7];
    const int*   sos    = (const int*)d_in[8];

    // Workspace layout (all 256B aligned):
    //   whh_pk  : 2 MB   (128 tiles x 16 kc x 32 lanes x 32 B, bf16)
    //   wih_pk  : 2 MB
    //   wpj_pk  : 112 KB (7 tiles), region padded to 128 KB
    //   xbuf    : 201 x 16 x 512 bf16  = 3,293,184 B
    //   xgbuf   : 201 x 128 x 32 x 8 f32 = 26,345,472 B
    char* ws = (char*)d_ws;
    unsigned short* whh_pk = (unsigned short*)(ws);
    unsigned short* wih_pk = (unsigned short*)(ws + (2u << 20));
    unsigned short* wpj_pk = (unsigned short*)(ws + (4u << 20));
    unsigned short* xbuf   = (unsigned short*)(ws + (4u << 20) + 131072);
    float*          xgbuf  = (float*)(ws + (4u << 20) + 131072 + 3293184);

    const int whalves = NTIL * NKC * 32 * 16;   // 1,048,576
    const int phalves = VTILES * NKC * 32 * 16; // 57,344
    pack_w<<<(whalves + 255) / 256, 256, 0, stream>>>(w_hh, whh_pk, G4H, whalves);
    pack_w<<<(whalves + 255) / 256, 256, 0, stream>>>(w_ih, wih_pk, G4H, whalves);
    pack_w<<<(phalves + 255) / 256, 256, 0, stream>>>(w_proj, wpj_pk, 100, phalves);

    lstm_main<<<1, NTHREADS, 0, stream>>>(feat, embed, sos, b_ih, b_hh,
                                          wih_pk, whh_pk, xbuf, xgbuf);

    proj_kern<<<dim3(TLEN, VTILES), 32, 0, stream>>>(xbuf, wpj_pk, b_proj,
                                                     (float*)d_out);
}
